// EnInvGraphConv_73349451481841
// MI455X (gfx1250) — compile-verified, matrix operand order checked by reading, not run on hardware
//
#include <hip/hip_runtime.h>
#include <hip/hip_bf16.h>
#include <math.h>

typedef __attribute__((ext_vector_type(16))) _Float16 v16h;
typedef __attribute__((ext_vector_type(8)))  float    v8f;

#define NB   2
#define NN   4096
#define DIM  128
#define KNB  32
#define CF   16
#define EIN  257     // edge input dim
#define EINP 288     // padded to 9*32
#define EH   514     // edge hidden dim
#define EHNP 528     // padded out dim of layer1 (33*16)
#define EHP  544     // padded K dim for layer2 (17*32), also LDS stride of H
#define NIN  144     // node input dim
#define NINP 160     // padded to 5*32
#define NH   256     // node hidden

__device__ __forceinline__ float gelu_exact(float x) {
    return 0.5f * x * (1.0f + erff(x * 0.7071067811865476f));
}

// Load one 16x32 f16 WMMA operand fragment (A by row, B by column — symmetric).
// Per ISA 7.12.2: lane holds row (lane&15); VGPR v holds k = 2v,2v+1 (v<4) or
// 16+2(v-4),.. (v>=4); lanes>=16 add +8. Two contiguous 16B runs per lane.
__device__ __forceinline__ v16h load_frag(const _Float16* base, int ld, int lane, int k0) {
    const _Float16* p = base + (lane & 15) * ld + k0 + ((lane >> 4) << 3);
    v16h r;
#pragma unroll
    for (int v = 0; v < 8; ++v) {
        int k = (v < 4) ? (2 * v) : (8 + 2 * v);   // 0,2,4,6,16,18,20,22
        r[2 * v]     = p[k];
        r[2 * v + 1] = p[k + 1];
    }
    return r;
}

#define WMMA_F16(acc, a, b) \
    __builtin_amdgcn_wmma_f32_16x16x32_f16(false, (a), false, (b), (short)0, (acc), false, false)

// ---------------------------------------------------------------------------
// Kernel 1: cast h -> f16; pad weights to f16 WMMA panels (zero padding so
// padded lanes are exact no-ops: GELU(0)=0 and padded weight cols are 0).
// ---------------------------------------------------------------------------
__global__ void prep_kernel(const float* __restrict__ h,
                            const float* __restrict__ We1, const float* __restrict__ be1,
                            const float* __restrict__ We2,
                            const float* __restrict__ Wn1, const float* __restrict__ Wn2,
                            _Float16* __restrict__ h16, _Float16* __restrict__ We1p,
                            float* __restrict__ be1p, _Float16* __restrict__ We2p,
                            _Float16* __restrict__ Wn1p, _Float16* __restrict__ Wn2p) {
    int tid = blockIdx.x * blockDim.x + threadIdx.x;
    int nth = gridDim.x * blockDim.x;
    for (int i = tid; i < NB * NN * DIM; i += nth) h16[i] = (_Float16)h[i];
    for (int i = tid; i < EHNP * EINP; i += nth) {
        int n = i / EINP, k = i - n * EINP;
        We1p[i] = (n < EH && k < EIN) ? (_Float16)We1[n * EIN + k] : (_Float16)0.f;
    }
    for (int i = tid; i < EHNP; i += nth) be1p[i] = (i < EH) ? be1[i] : 0.f;
    for (int i = tid; i < CF * EHP; i += nth) {
        int n = i / EHP, k = i - n * EHP;
        We2p[i] = (k < EH) ? (_Float16)We2[n * EH + k] : (_Float16)0.f;
    }
    for (int i = tid; i < NH * NINP; i += nth) {
        int n = i / NINP, k = i - n * NINP;
        Wn1p[i] = (k < NIN) ? (_Float16)Wn1[n * NIN + k] : (_Float16)0.f;
    }
    for (int i = tid; i < DIM * NH; i += nth) Wn2p[i] = (_Float16)Wn2[i];
}

// ---------------------------------------------------------------------------
// Kernel 2: top-K (K=32) smallest squared distances per node (unsorted set;
// order irrelevant since m_i sums over neighbors). One thread per node.
// ---------------------------------------------------------------------------
__global__ void topk_kernel(const float* __restrict__ x,
                            int* __restrict__ nbr_idx, float* __restrict__ nbr_dist) {
    __shared__ float xs[128 * 3];
    int node = blockIdx.x * 128 + threadIdx.x;          // block spans one batch
    int b = node >> 12;
    const float* xb = x + (size_t)b * NN * 3;
    float xi0 = x[node * 3 + 0], xi1 = x[node * 3 + 1], xi2 = x[node * 3 + 2];
    float dk[KNB];
    int   ik[KNB];
    int cnt = 0, wpos = 0;
    float wmax = -1.f;
    for (int t0 = 0; t0 < NN; t0 += 128) {
        __syncthreads();
        int j = t0 + threadIdx.x;
        xs[threadIdx.x * 3 + 0] = xb[j * 3 + 0];
        xs[threadIdx.x * 3 + 1] = xb[j * 3 + 1];
        xs[threadIdx.x * 3 + 2] = xb[j * 3 + 2];
        __syncthreads();
        for (int jj = 0; jj < 128; ++jj) {
            float dx = xi0 - xs[jj * 3 + 0];
            float dy = xi1 - xs[jj * 3 + 1];
            float dz = xi2 - xs[jj * 3 + 2];
            float dd = dx * dx + dy * dy + dz * dz;
            if (cnt < KNB) {
                dk[cnt] = dd; ik[cnt] = t0 + jj;
                if (dd > wmax) { wmax = dd; wpos = cnt; }
                ++cnt;
            } else if (dd < wmax) {
                dk[wpos] = dd; ik[wpos] = t0 + jj;
                wmax = dk[0]; wpos = 0;
#pragma unroll
                for (int t = 1; t < KNB; ++t)
                    if (dk[t] > wmax) { wmax = dk[t]; wpos = t; }
            }
        }
    }
#pragma unroll
    for (int t = 0; t < KNB; ++t) {
        nbr_idx[node * KNB + t]  = ik[t];
        nbr_dist[node * KNB + t] = dk[t];
    }
}

// ---------------------------------------------------------------------------
// Kernel 3: edge MLP + neighbor-sum. One block (8 waves) per node.
// layer1: mt outer, A-fragments (9) cached in registers (72 VGPRs, bounded —
// avoids the compiler's 18-fragment hoist that spilled to scratch); B streams
// from L2, reused across ~4 N-tiles per A fragment.
// ---------------------------------------------------------------------------
__global__ void __launch_bounds__(256, 1)
edge_kernel(const _Float16* __restrict__ h16,
            const _Float16* __restrict__ We1p, const float* __restrict__ be1p,
            const _Float16* __restrict__ We2p, const float* __restrict__ be2,
            const int* __restrict__ nbr_idx, const float* __restrict__ nbr_dist,
            float* __restrict__ m_i) {
    __shared__ _Float16 A1[KNB * EINP];   // 18 KB
    __shared__ _Float16 H[KNB * EHP];     // 34 KB
    __shared__ float m_acc[CF];
    int node = blockIdx.x;
    int b = node >> 12;
    int tid = threadIdx.x;
    int lane = tid & 31, wave = tid >> 5;
    const _Float16* hi = h16 + (size_t)node * DIM;

    for (int idx = tid; idx < KNB * EINP; idx += 256) {
        int row = idx / EINP, col = idx - row * EINP;
        _Float16 v;
        if (col < DIM) v = hi[col];
        else if (col < 2 * DIM) {
            int j = nbr_idx[node * KNB + row];
            v = h16[((size_t)(b << 12) + j) * DIM + (col - DIM)];
        } else if (col == 2 * DIM) v = (_Float16)nbr_dist[node * KNB + row];
        else v = (_Float16)0.f;
        A1[idx] = v;
    }
    // zero the H tail columns (528..543) so layer2's last K-tile sees no NaNs
    for (int idx = tid; idx < KNB * 16; idx += 256)
        H[(idx >> 4) * EHP + EHNP + (idx & 15)] = (_Float16)0.f;
    if (tid < CF) m_acc[tid] = 0.f;
    __syncthreads();

    int n15 = lane & 15;
    // layer1: waves split the 33 N-tiles; one M-tile's A fragments in regs
    for (int mt = 0; mt < 2; ++mt) {
        v16h afr[9];
#pragma unroll
        for (int kt = 0; kt < 9; ++kt)
            afr[kt] = load_frag(A1 + mt * 16 * EINP, EINP, lane, kt * 32);
        int mb = mt * 16 + ((lane >> 4) << 3);
        for (int nt = wave; nt < 33; nt += 8) {
            const _Float16* Bp = We1p + nt * 16 * EINP;
            v8f acc = {};
#pragma unroll
            for (int kt = 0; kt < 9; ++kt) {
                v16h bf = load_frag(Bp, EINP, lane, kt * 32);
                acc = WMMA_F16(acc, afr[kt], bf);
            }
            float bias = be1p[nt * 16 + n15];
#pragma unroll
            for (int r = 0; r < 8; ++r)
                H[(mb + r) * EHP + nt * 16 + n15] = (_Float16)gelu_exact(acc[r] + bias);
        }
    }
    __syncthreads();

    // layer2: waves 0,1 take one M-tile each; N = 16 (cf) single tile
    if (wave < 2) {
        int mt = wave;
        v8f acc = {};
        for (int kt = 0; kt < 17; ++kt) {
            v16h a  = load_frag(H + mt * 16 * EHP, EHP, lane, kt * 32);
            v16h bf = load_frag(We2p, EHP, lane, kt * 32);
            acc = WMMA_F16(acc, a, bf);
        }
        float bias = be2[n15];
        float s = 0.f;
#pragma unroll
        for (int r = 0; r < 8; ++r) s += gelu_exact(acc[r] + bias);
        s += __shfl_xor(s, 16, 32);                 // combine row halves (same col)
        if (lane < 16) atomicAdd(&m_acc[n15], s);   // combine the two M-tiles
    }
    __syncthreads();
    if (tid < CF) m_i[node * CF + tid] = m_acc[tid];
}

// ---------------------------------------------------------------------------
// Kernel 4: node MLP + residual. 128 threads = 4 waves; each wave owns 16
// nodes: [h|m_i|0pad] (16x160) -> 256 (GELU) -> 128, + bn2 + f32 h residual.
// A fragments cached in registers for both layers.
// ---------------------------------------------------------------------------
__global__ void __launch_bounds__(128, 1)
node_kernel(const _Float16* __restrict__ h16, const float* __restrict__ h,
            const float* __restrict__ m_i,
            const _Float16* __restrict__ Wn1p, const float* __restrict__ bn1,
            const _Float16* __restrict__ Wn2p, const float* __restrict__ bn2,
            float* __restrict__ out) {
    __shared__ _Float16 An[4][16 * NINP];   // 20 KB
    __shared__ _Float16 Hn[4][16 * NH];     // 32 KB
    int tid = threadIdx.x, lane = tid & 31, wave = tid >> 5;
    int base = (blockIdx.x * 4 + wave) * 16;
    _Float16* A  = An[wave];
    _Float16* Hh = Hn[wave];

    for (int idx = lane; idx < 16 * NINP; idx += 32) {
        int row = idx / NINP, col = idx - row * NINP;
        int node = base + row;
        _Float16 v;
        if (col < DIM)      v = h16[(size_t)node * DIM + col];
        else if (col < NIN) v = (_Float16)m_i[node * CF + (col - DIM)];
        else                v = (_Float16)0.f;
        A[idx] = v;
    }
    __syncthreads();

    int n15 = lane & 15;
    int mb = (lane >> 4) << 3;
    {
        v16h afr[5];
#pragma unroll
        for (int kt = 0; kt < 5; ++kt) afr[kt] = load_frag(A, NINP, lane, kt * 32);
        for (int nt = 0; nt < 16; ++nt) {
            v8f acc = {};
#pragma unroll
            for (int kt = 0; kt < 5; ++kt) {
                v16h bf = load_frag(Wn1p + nt * 16 * NINP, NINP, lane, kt * 32);
                acc = WMMA_F16(acc, afr[kt], bf);
            }
            float bias = bn1[nt * 16 + n15];
#pragma unroll
            for (int r = 0; r < 8; ++r)
                Hh[(mb + r) * NH + nt * 16 + n15] = (_Float16)gelu_exact(acc[r] + bias);
        }
    }
    __syncthreads();

    {
        v16h afr[8];
#pragma unroll
        for (int kt = 0; kt < 8; ++kt) afr[kt] = load_frag(Hh, NH, lane, kt * 32);
        for (int nt = 0; nt < 8; ++nt) {
            v8f acc = {};
#pragma unroll
            for (int kt = 0; kt < 8; ++kt) {
                v16h bf = load_frag(Wn2p + nt * 16 * NH, NH, lane, kt * 32);
                acc = WMMA_F16(acc, afr[kt], bf);
            }
            int col = nt * 16 + n15;
            float bias = bn2[col];
#pragma unroll
            for (int r = 0; r < 8; ++r) {
                int node = base + mb + r;
                out[(size_t)node * DIM + col] = acc[r] + bias + h[(size_t)node * DIM + col];
            }
        }
    }
}

// ---------------------------------------------------------------------------
extern "C" void kernel_launch(void* const* d_in, const int* in_sizes, int n_in,
                              void* d_out, int out_size, void* d_ws, size_t ws_size,
                              hipStream_t stream) {
    (void)in_sizes; (void)n_in; (void)out_size; (void)ws_size;
    const float* h   = (const float*)d_in[0];
    const float* x   = (const float*)d_in[1];
    const float* We1 = (const float*)d_in[2];
    const float* be1 = (const float*)d_in[3];
    const float* We2 = (const float*)d_in[4];
    const float* be2 = (const float*)d_in[5];
    const float* Wn1 = (const float*)d_in[6];
    const float* bn1 = (const float*)d_in[7];
    const float* Wn2 = (const float*)d_in[8];
    const float* bn2 = (const float*)d_in[9];
    float* out = (float*)d_out;

    char* ws = (char*)d_ws;
    size_t off = 0;
    auto carve = [&](size_t bytes) -> void* {
        void* p = ws + off;
        off = (off + bytes + 255) & ~(size_t)255;
        return p;
    };
    _Float16* h16   = (_Float16*)carve((size_t)NB * NN * DIM * 2);
    _Float16* We1p  = (_Float16*)carve((size_t)EHNP * EINP * 2);
    _Float16* We2p  = (_Float16*)carve((size_t)CF * EHP * 2);
    _Float16* Wn1p  = (_Float16*)carve((size_t)NH * NINP * 2);
    _Float16* Wn2p  = (_Float16*)carve((size_t)DIM * NH * 2);
    float*    be1p  = (float*)carve((size_t)EHNP * 4);
    int*      nbri  = (int*)carve((size_t)NB * NN * KNB * 4);
    float*    nbrd  = (float*)carve((size_t)NB * NN * KNB * 4);
    float*    m_i   = (float*)carve((size_t)NB * NN * CF * 4);

    prep_kernel<<<2048, 256, 0, stream>>>(h, We1, be1, We2, Wn1, Wn2,
                                          h16, We1p, be1p, We2p, Wn1p, Wn2p);
    topk_kernel<<<(NB * NN) / 128, 128, 0, stream>>>(x, nbri, nbrd);
    edge_kernel<<<NB * NN, 256, 0, stream>>>(h16, We1p, be1p, We2p, be2,
                                             nbri, nbrd, m_i);
    node_kernel<<<(NB * NN) / 64, 128, 0, stream>>>(h16, h, m_i, Wn1p, bn1,
                                                    Wn2p, bn2, out);
}
